// GCNNet_19679540150778
// MI455X (gfx1250) — matrix operand order repaired and checked by reading, not compiled, not run on hardware
//
#include <hip/hip_runtime.h>

// GCN (3x GCNConv + linear head + log_softmax) for MI455X (gfx1250, wave32).
//
// Roofline: edge gather/scatter dominates (~5.2 GB of L2-resident traffic +
// 614M f32 atomics); the 51MB aggregation buffer fits in the 192MB L2 so
// global_atomic_add_f32 resolves at L2. GEMMs (9.8 GFLOP) run on
// v_wmma_f32_16x16x32_f16 with an f16 hi/lo split for fp32-class accuracy.
// Scatter uses 1 lane per 4 features: b128 gather + 4 atomics per lane,
// cutting redundant index/norm broadcast loads 4x vs 1 lane per feature.

typedef __attribute__((ext_vector_type(16))) _Float16 v16h;
typedef __attribute__((ext_vector_type(8)))  float    v8f;

#define HID    128
#define NCLS   10
#define NLAYERS 3
#define XCATW  (NLAYERS * HID)   // 384

__device__ __forceinline__ v8f wmma_f16(v16h a, v16h b, v8f c) {
  // (neg_a, A, neg_b, B, c_mod, C, reuse_a, reuse_b)
  return __builtin_amdgcn_wmma_f32_16x16x32_f16(false, a, false, b, (short)0, c,
                                                false, false);
}

// ---------------- degree / normalization prep ----------------
__global__ void k_deg_init(float* __restrict__ deg, int n) {
  int i = blockIdx.x * blockDim.x + threadIdx.x;
  if (i < n) deg[i] = 1.0f;  // self loop
}

__global__ void k_deg_accum(const int* __restrict__ ei, float* __restrict__ deg, int nE) {
  int e = blockIdx.x * blockDim.x + threadIdx.x;
  if (e < nE) atomicAdd(&deg[ei[nE + e]], 1.0f);  // deg over dst
}

__global__ void k_dinv(float* __restrict__ deg, int n) {
  int i = blockIdx.x * blockDim.x + threadIdx.x;
  if (i < n) deg[i] = rsqrtf(deg[i]);
}

__global__ void k_norm(const int* __restrict__ ei, const float* __restrict__ dinv,
                       float* __restrict__ norm, int nE) {
  int e = blockIdx.x * blockDim.x + threadIdx.x;
  if (e < nE) norm[e] = dinv[ei[e]] * dinv[ei[nE + e]];
}

// ---------------- weight pre-swizzle into WMMA B-fragment layout ----------------
// Layout: [tile t(0..7)][kchunk kc(0..3)][lane(0..31)][elem j(0..15)]
// B 32x16 (KxN) 16-bit mapping: n = t*16 + (lane&15); K = kc*32 + (lane>>4)*8 + (j&7) + (j>>3)*16
__global__ void k_prep_w(const float* __restrict__ W, _Float16* __restrict__ bhi,
                         _Float16* __restrict__ blo) {
  for (int i = threadIdx.x; i < 8 * 4 * 32 * 16; i += blockDim.x) {
    int j    = i & 15;
    int lane = (i >> 4) & 31;
    int kc   = (i >> 9) & 3;
    int t    = i >> 11;
    int nn   = t * 16 + (lane & 15);
    int K    = kc * 32 + (lane >> 4) * 8 + (j & 7) + (j >> 3) * 16;
    float v  = W[K * HID + nn];
    _Float16 hi = (_Float16)v;
    bhi[i] = hi;
    blo[i] = (_Float16)(v - (float)hi);
  }
}

// ---------------- WMMA GEMM: h[N,128] = xin[N,128] @ W[128,128] ----------------
// One wave computes a 16x128 output tile. K=128 in 4 chunks of 32.
// Split-f16: x = ahi + alo  =>  acc += ahi*Whi + alo*Whi + ahi*Wlo
__global__ void k_gemm(const float* __restrict__ xin, int xstride,
                       const _Float16* __restrict__ bhi, const _Float16* __restrict__ blo,
                       float* __restrict__ hout, int n) {
  int wave = blockIdx.x * (blockDim.x >> 5) + (threadIdx.x >> 5);
  int lane = threadIdx.x & 31;
  int m0 = wave << 4;
  if (m0 >= n) return;  // wave-uniform: whole wave exits or stays (EXEC all-1s for WMMA)

  int mrow = lane & 15;
  int kb   = (lane >> 4) << 3;  // 0 or 8 (A-frag K interleave per ISA table)
  const float* xrow = xin + (size_t)(m0 + mrow) * xstride;

  v8f zero = {};
  v8f acc[8];
#pragma unroll
  for (int t = 0; t < 8; ++t) acc[t] = zero;

#pragma unroll
  for (int kc = 0; kc < 4; ++kc) {
    // A 16x32 f16 fragment, lane holds: K = kc*32 + kb + {0..7} and + {16..23}
    const float* ab = xrow + kc * 32 + kb;
    float xv[16];
#pragma unroll
    for (int j = 0; j < 8; ++j) { xv[j] = ab[j]; xv[8 + j] = ab[16 + j]; }
    v16h ahi, alo;
#pragma unroll
    for (int j = 0; j < 16; ++j) {
      _Float16 hi = (_Float16)xv[j];
      ahi[j] = hi;
      alo[j] = (_Float16)(xv[j] - (float)hi);
    }
#pragma unroll
    for (int t = 0; t < 8; ++t) {
      v16h bh = *(const v16h*)(bhi + (((t * 4 + kc) * 32 + lane) << 4));
      v16h bw = *(const v16h*)(blo + (((t * 4 + kc) * 32 + lane) << 4));
      acc[t] = wmma_f16(ahi, bh, acc[t]);
      acc[t] = wmma_f16(alo, bh, acc[t]);
      acc[t] = wmma_f16(ahi, bw, acc[t]);
    }
  }

  // D layout: lane&15 = column, VGPR r -> row r + 8*(lane>=16)
  int mo = (lane >> 4) << 3;
  int nc = lane & 15;
#pragma unroll
  for (int t = 0; t < 8; ++t)
#pragma unroll
    for (int r = 0; r < 8; ++r)
      hout[(size_t)(m0 + mo + r) * HID + t * 16 + nc] = acc[t][r];
}

// ---------------- aggregation ----------------
// Self-loop term initializes the xcat slice (also serves as zero-init).
// 1 thread per 4 features -> b128 load + b128 store.
__global__ void k_selfinit(const float* __restrict__ h, const float* __restrict__ dinv,
                           float* __restrict__ xcat, int col0, int n) {
  int i = blockIdx.x * blockDim.x + threadIdx.x;
  if (i >= n * (HID / 4)) return;
  int node = i >> 5, f = (i & 31) << 2;
  float d = dinv[node];
  float s = d * d;
  float4 hv = *(const float4*)(h + (size_t)node * HID + f);
  float4 o;
  o.x = hv.x * s; o.y = hv.y * s; o.z = hv.z * s; o.w = hv.w * s;
  *(float4*)(xcat + (size_t)node * XCATW + col0 + f) = o;
}

// 1 lane per (edge, 4 features): 32 lanes cover one edge's 128 features.
// b128 gather of h[src] + 4 contiguous f32 atomics into L2-resident xcat;
// src/dst/norm loads broadcast within the wave.
__global__ void k_scatter(const int* __restrict__ ei, const float* __restrict__ norm,
                          const float* __restrict__ h, float* __restrict__ xcat,
                          int col0, int nE) {
  long long tid = (long long)blockIdx.x * blockDim.x + threadIdx.x;
  int e = (int)(tid >> 5);
  if (e >= nE) return;
  int f = ((int)tid & 31) << 2;
  int s = ei[e];
  int d = ei[nE + e];
  float nrm = norm[e];
  float4 hv = *(const float4*)(h + (size_t)s * HID + f);
  float* dst = xcat + (size_t)d * XCATW + col0 + f;
  atomicAdd(dst + 0, hv.x * nrm);
  atomicAdd(dst + 1, hv.y * nrm);
  atomicAdd(dst + 2, hv.z * nrm);
  atomicAdd(dst + 3, hv.w * nrm);
}

__global__ void k_biasrelu(float* __restrict__ xcat, const float* __restrict__ bc,
                           int col0, int n) {
  int i = blockIdx.x * blockDim.x + threadIdx.x;
  if (i >= n * (HID / 4)) return;
  int node = i >> 5, f = (i & 31) << 2;
  float4 b = *(const float4*)(bc + f);
  float4* p = (float4*)(xcat + (size_t)node * XCATW + col0 + f);
  float4 v = *p;
  v.x = fmaxf(v.x + b.x, 0.f);
  v.y = fmaxf(v.y + b.y, 0.f);
  v.z = fmaxf(v.z + b.z, 0.f);
  v.w = fmaxf(v.w + b.w, 0.f);
  *p = v;
}

// ---------------- head: logits = xcat @ Wl + bl ----------------
__global__ void k_linear(const float* __restrict__ xcat, const float* __restrict__ Wl,
                         const float* __restrict__ bl, float* __restrict__ out, int n) {
  int tid = blockIdx.x * blockDim.x + threadIdx.x;
  int node = tid >> 4, c = tid & 15;
  if (node >= n || c >= NCLS) return;
  const float* xr = xcat + (size_t)node * XCATW;
  float acc = bl[c];
#pragma unroll 2
  for (int k = 0; k < XCATW; k += 4) {
    float4 xv = *(const float4*)(xr + k);
    acc += xv.x * Wl[(k + 0) * NCLS + c];
    acc += xv.y * Wl[(k + 1) * NCLS + c];
    acc += xv.z * Wl[(k + 2) * NCLS + c];
    acc += xv.w * Wl[(k + 3) * NCLS + c];
  }
  out[(size_t)node * NCLS + c] = acc;
}

__global__ void k_logsoftmax(float* __restrict__ out, int n) {
  int node = blockIdx.x * blockDim.x + threadIdx.x;
  if (node >= n) return;
  float* row = out + (size_t)node * NCLS;
  float l[NCLS];
  float m = -1e30f;
#pragma unroll
  for (int c = 0; c < NCLS; ++c) { l[c] = row[c]; m = fmaxf(m, l[c]); }
  float s = 0.f;
#pragma unroll
  for (int c = 0; c < NCLS; ++c) s += expf(l[c] - m);
  float lse = m + logf(s);
#pragma unroll
  for (int c = 0; c < NCLS; ++c) row[c] = l[c] - lse;
}

extern "C" void kernel_launch(void* const* d_in, const int* in_sizes, int n_in,
                              void* d_out, int out_size, void* d_ws, size_t ws_size,
                              hipStream_t stream) {
  const float* feat = (const float*)d_in[0];
  const int*   ei   = (const int*)d_in[1];
  const float* Wc   = (const float*)d_in[2];
  const float* bc   = (const float*)d_in[3];
  const float* Wl   = (const float*)d_in[4];
  const float* bl   = (const float*)d_in[5];
  float* out = (float*)d_out;

  const int n  = in_sizes[0] / HID;  // 100000
  const int nE = in_sizes[1] / 2;    // 1600000

  // workspace carve-out (~212 MB)
  char* ws = (char*)d_ws;
  size_t off = 0;
  float* xcat = (float*)(ws + off); off += (size_t)n * XCATW * sizeof(float);
  float* h    = (float*)(ws + off); off += (size_t)n * HID * sizeof(float);
  float* dinv = (float*)(ws + off); off += (((size_t)n * sizeof(float)) + 255) & ~(size_t)255;
  float* norm = (float*)(ws + off); off += (((size_t)nE * sizeof(float)) + 255) & ~(size_t)255;
  _Float16* bhi = (_Float16*)(ws + off); off += 16384 * sizeof(_Float16);
  _Float16* blo = (_Float16*)(ws + off); off += 16384 * sizeof(_Float16);
  (void)ws_size; (void)n_in; (void)out_size;

  const int B = 256;
  k_deg_init <<<(n + B - 1) / B,  B, 0, stream>>>(dinv, n);
  k_deg_accum<<<(nE + B - 1) / B, B, 0, stream>>>(ei, dinv, nE);
  k_dinv     <<<(n + B - 1) / B,  B, 0, stream>>>(dinv, n);
  k_norm     <<<(nE + B - 1) / B, B, 0, stream>>>(ei, dinv, norm, nE);

  for (int layer = 0; layer < NLAYERS; ++layer) {
    const float* xin = (layer == 0) ? feat : (xcat + (size_t)(layer - 1) * HID);
    int xstride = (layer == 0) ? HID : XCATW;

    k_prep_w<<<1, 256, 0, stream>>>(Wc + (size_t)layer * HID * HID, bhi, blo);

    int waves  = (n + 15) / 16;            // 6250 tiles of 16 rows
    int gblocks = (waves + 7) / 8;         // 8 waves per 256-thread block
    k_gemm<<<gblocks, 256, 0, stream>>>(xin, xstride, bhi, blo, h, n);

    int nf4 = n * (HID / 4);               // 3.2M float4 slots
    k_selfinit<<<(nf4 + B - 1) / B, B, 0, stream>>>(h, dinv, xcat, layer * HID, n);

    long long sthreads = (long long)nE * 32;  // 51.2M (1 lane = 4 features)
    unsigned sblocks = (unsigned)((sthreads + B - 1) / B);
    k_scatter<<<sblocks, B, 0, stream>>>(ei, norm, h, xcat, layer * HID, nE);

    k_biasrelu<<<(nf4 + B - 1) / B, B, 0, stream>>>(xcat, bc + (size_t)layer * HID,
                                                    layer * HID, n);
  }

  int lthreads = n * 16;  // classes padded to 16
  k_linear    <<<(lthreads + B - 1) / B, B, 0, stream>>>(xcat, Wl, bl, out, n);
  k_logsoftmax<<<(n + B - 1) / B,        B, 0, stream>>>(out, n);
}
